// SampleNetDropout_87866440941654
// MI455X (gfx1250) — matrix operand order repaired and checked by reading, not compiled, not run on hardware
//
#include <hip/hip_runtime.h>
#include <hip/hip_bf16.h>

typedef __attribute__((ext_vector_type(16))) _Float16 v16h;
typedef __attribute__((ext_vector_type(8)))  _Float16 v8h;
typedef __attribute__((ext_vector_type(8)))  float    v8f;

union V16 { v16h v; v8h h[2]; };

// ---- WMMA helper: D = A(16x32 f16) * B(32x16 f16) + C(16x16 f32) ----
#define WMMA(a, b, c) __builtin_amdgcn_wmma_f32_16x16x32_f16( \
    false, (a), false, (b), (short)0, (c), false, false)

// A operand (16x32 f16 tile) from f16 LDS, row-major, leading dim `ld` (halfs).
// ISA layout: lanes 0-15 row M=lane, K in {0..7,16..23}; lanes 16-31 row M=lane-16,
// K in {8..15,24..31}; element pairs per VGPR are consecutive K.
__device__ __forceinline__ v16h load_A_lds(const _Float16* base, int ld) {
  const int lane = threadIdx.x & 31;
  const _Float16* p = base + (lane & 15) * ld + ((lane >> 4) << 3);
  V16 u;
  u.h[0] = *(const v8h*)(p);        // K = kh .. kh+7
  u.h[1] = *(const v8h*)(p + 16);   // K = kh+16 .. kh+23
  return u.v;
}

// A operand from row-major f32 memory (global OR LDS), converted to f16.
__device__ __forceinline__ v16h load_A_f32(const float* base, int ld) {
  const int lane = threadIdx.x & 31;
  const float* p = base + (lane & 15) * ld + ((lane >> 4) << 3);
  v16h a;
#pragma unroll
  for (int i = 0; i < 8; ++i) {
    a[i]     = (_Float16)p[i];
    a[i + 8] = (_Float16)p[i + 16];
  }
  return a;
}

// B operand (32x16 f16 tile) from transposed weights in LDS: wt[n*ld + k] = W[k][n].
// Lanes 0-15: column N=n0+lane, K=k0+{0..7,16..23}; lanes 16-31: K=k0+{8..15,24..31}.
__device__ __forceinline__ v16h load_B_lds(const _Float16* wt, int ld, int n0, int k0) {
  const int lane = threadIdx.x & 31;
  const _Float16* p = wt + (n0 + (lane & 15)) * ld + k0 + ((lane >> 4) << 3);
  V16 u;
  u.h[0] = *(const v8h*)(p);
  u.h[1] = *(const v8h*)(p + 16);
  return u.v;
}

// Branchless ELU on the hardware v_exp_f32 path (no expm1 polynomial, no branches).
__device__ __forceinline__ float elu1(float x) {
  float e = __expf(x) - 1.0f;
  return x > 0.0f ? x : e;
}

// C/D layout: VGPR v, lanes 0-15 -> (M=v, N=lane); lanes 16-31 -> (M=8+v, N=lane-16).
__device__ __forceinline__ void store_C_elu(_Float16* dst, int ld, v8f c, const float* bias) {
  const int lane = threadIdx.x & 31;
  const int col  = lane & 15;
  const int r0   = (lane >> 4) << 3;
  const float b  = bias[col];
#pragma unroll
  for (int v = 0; v < 8; ++v)
    dst[(r0 + v) * ld + col] = (_Float16)elu1(c[v] + b);
}

__device__ __forceinline__ void store_C_elu_addg(_Float16* dst, int ld, v8f c,
                                                 const float* bias,
                                                 const _Float16* g, int ldg) {
  const int lane = threadIdx.x & 31;
  const int col  = lane & 15;
  const int r0   = (lane >> 4) << 3;
  const float b  = bias[col];
#pragma unroll
  for (int v = 0; v < 8; ++v) {
    float val = elu1(c[v] + b) + (float)g[(r0 + v) * ldg + col];
    dst[(r0 + v) * ld + col] = (_Float16)val;
  }
}

// CDNA5 async copy: 16 bytes global -> LDS, tracked by ASYNCcnt (no VGPR data path).
// Shared immediate offset applies to both the global and LDS addresses.
#define ASYNC_B128(ldsb, gp, imm)                                         \
  asm volatile("global_load_async_to_lds_b128 %0, %1, off offset:%c2"     \
               :: "v"(ldsb), "v"(gp), "i"(imm) : "memory")

// ============================================================================
// Kernel 1: per-edge MLP chain + gather + contiguous segment reduction.
// One block = one sample (256 edge rows). 256 threads = 8 waves, 32 rows/wave.
// ============================================================================
__global__ __launch_bounds__(256)
void k1_edge(const float* __restrict__ xin,        // sample_node_feature [E,64]
             const int*   __restrict__ nid,        // sample_node_id [E,2]
             const float* __restrict__ gtab,       // graph_out [N_NODES,32]
             const float* w1, const float* b1,     // nn_w1/b1 (64x32)
             const float* w2, const float* b2,     // nn_w2/b2 (32x32)
             const float* w3, const float* b3,     // g1_w1/b1 (32x32)
             const float* w4, const float* b4,     // g1_w2/b2 (32x32)
             float* __restrict__ xsum)             // [S,32] segment sums
{
  constexpr int LDXF = 68, LDH = 40, LDW1 = 72, LDW = 40;
  __shared__ __align__(16) float    Xf32[256 * LDXF];   // async-staged input rows
  __shared__ __align__(16) _Float16 Gs[256 * LDH];
  __shared__ __align__(16) _Float16 Hs[256 * LDH];
  __shared__ __align__(16) _Float16 WT1[32 * LDW1];
  __shared__ __align__(16) _Float16 WT2[32 * LDW];
  __shared__ __align__(16) _Float16 WT3[32 * LDW];
  __shared__ __align__(16) _Float16 WT4[32 * LDW];
  __shared__ float B1[32], B2[32], B3[32], B4[32];
  __shared__ float Wacc[8][32];

  const int t = threadIdx.x;
  const long long blk = blockIdx.x;

  // --- async-stage this sample's 256x64 f32 input rows into LDS (ASYNCcnt) ---
  {
    const float* xr = xin + (blk * 256 + t) * 64;          // 256 B per row
    const unsigned ldsb = (unsigned)(uintptr_t)(Xf32 + t * LDXF);
    ASYNC_B128(ldsb, xr, 0);   ASYNC_B128(ldsb, xr, 16);
    ASYNC_B128(ldsb, xr, 32);  ASYNC_B128(ldsb, xr, 48);
    ASYNC_B128(ldsb, xr, 64);  ASYNC_B128(ldsb, xr, 80);
    ASYNC_B128(ldsb, xr, 96);  ASYNC_B128(ldsb, xr, 112);
    ASYNC_B128(ldsb, xr, 128); ASYNC_B128(ldsb, xr, 144);
    ASYNC_B128(ldsb, xr, 160); ASYNC_B128(ldsb, xr, 176);
    ASYNC_B128(ldsb, xr, 192); ASYNC_B128(ldsb, xr, 208);
    ASYNC_B128(ldsb, xr, 224); ASYNC_B128(ldsb, xr, 240);
  }

  // --- stage weights (transposed, f16) + biases into LDS ---
  for (int i = t; i < 64 * 32; i += 256) {
    int k = i >> 5, n = i & 31;
    WT1[n * LDW1 + k] = (_Float16)w1[i];
  }
  for (int i = t; i < 32 * 32; i += 256) {
    int k = i >> 5, n = i & 31;
    WT2[n * LDW + k] = (_Float16)w2[i];
    WT3[n * LDW + k] = (_Float16)w3[i];
    WT4[n * LDW + k] = (_Float16)w4[i];
  }
  if (t < 32) { B1[t] = b1[t]; B2[t] = b2[t]; B3[t] = b3[t]; B4[t] = b4[t]; }

  // --- gathered graph rows (L2-resident 12.8 MB table) -> f16 LDS ---
  {
    const int node = nid[(blk * 256 + t) * 2 + 1];
    const float* gr = gtab + (long long)node * 32;
#pragma unroll
    for (int k = 0; k < 32; k += 4) {
      float4 f = *(const float4*)(gr + k);
      Gs[t * LDH + k + 0] = (_Float16)f.x;
      Gs[t * LDH + k + 1] = (_Float16)f.y;
      Gs[t * LDH + k + 2] = (_Float16)f.z;
      Gs[t * LDH + k + 3] = (_Float16)f.w;
    }
  }
  asm volatile("s_wait_asynccnt 0" ::: "memory");
  __syncthreads();

  const int wv = t >> 5;
  const int m0 = wv * 32;

  // ---- stage 1: Xf32[.,64] @ W1 + b1 -> elu -> Hs (A converted f32->f16 on read) ----
#pragma unroll
  for (int mt = 0; mt < 2; ++mt) {
    const int r0 = m0 + mt * 16;
    v16h a0 = load_A_f32(Xf32 + r0 * LDXF, LDXF);
    v16h a1 = load_A_f32(Xf32 + r0 * LDXF + 32, LDXF);
#pragma unroll
    for (int nt = 0; nt < 2; ++nt) {
      v8f c = {};
      c = WMMA(a0, load_B_lds(WT1, LDW1, nt * 16, 0),  c);
      c = WMMA(a1, load_B_lds(WT1, LDW1, nt * 16, 32), c);
      store_C_elu(Hs + r0 * LDH + nt * 16, LDH, c, B1 + nt * 16);
    }
  }
  // ---- stage 2: Hs @ W2 + b2 -> elu -> (+ gathered G) -> Hs ----
#pragma unroll
  for (int mt = 0; mt < 2; ++mt) {
    const int r0 = m0 + mt * 16;
    v16h a = load_A_lds(Hs + r0 * LDH, LDH);
#pragma unroll
    for (int nt = 0; nt < 2; ++nt) {
      v8f c = {};
      c = WMMA(a, load_B_lds(WT2, LDW, nt * 16, 0), c);
      store_C_elu_addg(Hs + r0 * LDH + nt * 16, LDH, c, B2 + nt * 16,
                       Gs + r0 * LDH + nt * 16, LDH);
    }
  }
  // ---- stage 3: Hs @ g1_w1 + b -> elu -> Hs ----
#pragma unroll
  for (int mt = 0; mt < 2; ++mt) {
    const int r0 = m0 + mt * 16;
    v16h a = load_A_lds(Hs + r0 * LDH, LDH);
#pragma unroll
    for (int nt = 0; nt < 2; ++nt) {
      v8f c = {};
      c = WMMA(a, load_B_lds(WT3, LDW, nt * 16, 0), c);
      store_C_elu(Hs + r0 * LDH + nt * 16, LDH, c, B3 + nt * 16);
    }
  }
  // ---- stage 4: Hs @ g1_w2 + b -> elu -> reduce 32 rows/wave (deterministic) ----
  {
    v16h a0 = load_A_lds(Hs + m0 * LDH, LDH);
    v16h a1 = load_A_lds(Hs + (m0 + 16) * LDH, LDH);
    const int lane = t & 31;
    const int col  = lane & 15;
#pragma unroll
    for (int nt = 0; nt < 2; ++nt) {
      v16h bop = load_B_lds(WT4, LDW, nt * 16, 0);
      v8f c0 = {}, c1 = {};
      c0 = WMMA(a0, bop, c0);
      c1 = WMMA(a1, bop, c1);
      const float bb = B4[nt * 16 + col];
      float s = 0.0f;
#pragma unroll
      for (int v = 0; v < 8; ++v) { s += elu1(c0[v] + bb); s += elu1(c1[v] + bb); }
      s += __shfl_xor(s, 16, 32);           // combine M-halves (lanes L, L+16)
      if (lane < 16) Wacc[wv][nt * 16 + col] = s;
    }
  }
  __syncthreads();
  if (t < 32) {
    float s = 0.0f;
#pragma unroll
    for (int w = 0; w < 8; ++w) s += Wacc[w][t];
    xsum[blk * 32 + t] = s;
  }
}

// ============================================================================
// Kernel 2: per-sample head. 32 blocks x 256 threads; 256 samples per block.
// out_feature = mlp2(sample_feature) (segment-mean is identity since
// sample_id == arange(S)); out_graph = mlp2(xsum); then o-MLP + sigmoid.
// ============================================================================
__global__ __launch_bounds__(256)
void k2_head(const float* __restrict__ xsum,       // [S,32]
             const float* __restrict__ sfeat,      // sample_feature [S,256]
             const float* g2w1, const float* g2b1, const float* g2w2, const float* g2b2,
             const float* fw1,  const float* fb1,  const float* fw2,  const float* fb2,
             const float* ow1,  const float* ob1,  const float* ow2,  const float* ob2,
             const float* ow3,  const float* ob3,
             float* __restrict__ out)              // [S,2]
{
  constexpr int LDC = 72, LDH = 40, LDF = 264, LDO1 = 72, LDW = 40;
  __shared__ __align__(16) _Float16 Cat[256 * LDC];   // [rows][0:32]=feat/h2, [32:64]=graph
  __shared__ __align__(16) _Float16 H[256 * LDH];
  __shared__ __align__(16) _Float16 FW1T[32 * LDF];
  __shared__ __align__(16) _Float16 OW1T[32 * LDO1];
  __shared__ __align__(16) _Float16 G2W1T[32 * LDW];
  __shared__ __align__(16) _Float16 G2W2T[32 * LDW];
  __shared__ __align__(16) _Float16 FW2T[32 * LDW];
  __shared__ __align__(16) _Float16 OW2T[32 * LDW];
  __shared__ float OW3[64];
  __shared__ float G2B1[32], G2B2[32], FB1[32], FB2[32], OB1[32], OB2[32], OB3[2];

  const int t = threadIdx.x;
  const long long s0 = (long long)blockIdx.x * 256;

  for (int i = t; i < 256 * 32; i += 256) {
    int k = i >> 5, n = i & 31;
    FW1T[n * LDF + k] = (_Float16)fw1[i];
  }
  for (int i = t; i < 64 * 32; i += 256) {
    int k = i >> 5, n = i & 31;
    OW1T[n * LDO1 + k] = (_Float16)ow1[i];
  }
  for (int i = t; i < 32 * 32; i += 256) {
    int k = i >> 5, n = i & 31;
    G2W1T[n * LDW + k] = (_Float16)g2w1[i];
    G2W2T[n * LDW + k] = (_Float16)g2w2[i];
    FW2T[n * LDW + k]  = (_Float16)fw2[i];
    OW2T[n * LDW + k]  = (_Float16)ow2[i];
  }
  if (t < 64) OW3[t] = ow3[t];
  if (t < 32) {
    G2B1[t] = g2b1[t]; G2B2[t] = g2b2[t];
    FB1[t]  = fb1[t];  FB2[t]  = fb2[t];
    OB1[t]  = ob1[t];  OB2[t]  = ob2[t];
  }
  if (t < 2) OB3[t] = ob3[t];
  __syncthreads();

  const int wv = t >> 5;
  const int m0 = wv * 32;

#pragma unroll
  for (int mt = 0; mt < 2; ++mt) {
    const int r0 = m0 + mt * 16;

    // f1: sample_feature[rows,256] @ f_w1 (8 K-chunks) -> elu -> H
    {
      v8f c0 = {}, c1 = {};
#pragma unroll
      for (int kc = 0; kc < 256; kc += 32) {
        v16h a = load_A_f32(sfeat + (s0 + r0) * 256 + kc, 256);
        c0 = WMMA(a, load_B_lds(FW1T, LDF, 0,  kc), c0);
        c1 = WMMA(a, load_B_lds(FW1T, LDF, 16, kc), c1);
      }
      store_C_elu(H + r0 * LDH + 0,  LDH, c0, FB1 + 0);
      store_C_elu(H + r0 * LDH + 16, LDH, c1, FB1 + 16);
    }
    // f2: H @ f_w2 -> elu -> Cat[:,0:32]
    {
      v16h a = load_A_lds(H + r0 * LDH, LDH);
#pragma unroll
      for (int nt = 0; nt < 2; ++nt) {
        v8f c = {};
        c = WMMA(a, load_B_lds(FW2T, LDW, nt * 16, 0), c);
        store_C_elu(Cat + r0 * LDC + nt * 16, LDC, c, FB2 + nt * 16);
      }
    }
    // g1: xsum[rows,32] @ g2_w1 -> elu -> H
    {
      v16h a = load_A_f32(xsum + (s0 + r0) * 32, 32);
#pragma unroll
      for (int nt = 0; nt < 2; ++nt) {
        v8f c = {};
        c = WMMA(a, load_B_lds(G2W1T, LDW, nt * 16, 0), c);
        store_C_elu(H + r0 * LDH + nt * 16, LDH, c, G2B1 + nt * 16);
      }
    }
    // g2: H @ g2_w2 -> elu -> Cat[:,32:64]
    {
      v16h a = load_A_lds(H + r0 * LDH, LDH);
#pragma unroll
      for (int nt = 0; nt < 2; ++nt) {
        v8f c = {};
        c = WMMA(a, load_B_lds(G2W2T, LDW, nt * 16, 0), c);
        store_C_elu(Cat + r0 * LDC + 32 + nt * 16, LDC, c, G2B2 + nt * 16);
      }
    }
    // o1: Cat[rows,64] @ o_w1 -> elu -> H
    {
      v16h a0 = load_A_lds(Cat + r0 * LDC, LDC);
      v16h a1 = load_A_lds(Cat + r0 * LDC + 32, LDC);
#pragma unroll
      for (int nt = 0; nt < 2; ++nt) {
        v8f c = {};
        c = WMMA(a0, load_B_lds(OW1T, LDO1, nt * 16, 0),  c);
        c = WMMA(a1, load_B_lds(OW1T, LDO1, nt * 16, 32), c);
        store_C_elu(H + r0 * LDH + nt * 16, LDH, c, OB1 + nt * 16);
      }
    }
    // o2: H @ o_w2 -> elu -> Cat[:,0:32] (reuse)
    {
      v16h a = load_A_lds(H + r0 * LDH, LDH);
#pragma unroll
      for (int nt = 0; nt < 2; ++nt) {
        v8f c = {};
        c = WMMA(a, load_B_lds(OW2T, LDW, nt * 16, 0), c);
        store_C_elu(Cat + r0 * LDC + nt * 16, LDC, c, OB2 + nt * 16);
      }
    }
  }

  // final: h2[32] @ o_w3[32,2] + b3 -> sigmoid (one row per lane)
  {
    const int lane = t & 31;
    const int r = m0 + lane;
    float a0 = OB3[0], a1 = OB3[1];
#pragma unroll
    for (int k = 0; k < 32; ++k) {
      float h = (float)Cat[r * LDC + k];
      a0 += h * OW3[2 * k];
      a1 += h * OW3[2 * k + 1];
    }
    out[(s0 + r) * 2 + 0] = 1.0f / (1.0f + __expf(-a0));
    out[(s0 + r) * 2 + 1] = 1.0f / (1.0f + __expf(-a1));
  }
}

// ============================================================================
extern "C" void kernel_launch(void* const* d_in, const int* in_sizes, int n_in,
                              void* d_out, int out_size, void* d_ws, size_t ws_size,
                              hipStream_t stream) {
  (void)in_sizes; (void)n_in; (void)out_size; (void)ws_size;
  const float* graph_out = (const float*)d_in[0];
  const int*   sn_id     = (const int*)d_in[1];
  const float* sn_feat   = (const float*)d_in[2];
  /* d_in[3] sample_id: identity arange, unused */
  const float* s_feat    = (const float*)d_in[4];
  const float* nn_w1 = (const float*)d_in[5],  *nn_b1 = (const float*)d_in[6];
  const float* nn_w2 = (const float*)d_in[7],  *nn_b2 = (const float*)d_in[8];
  const float* g1_w1 = (const float*)d_in[9],  *g1_b1 = (const float*)d_in[10];
  const float* g1_w2 = (const float*)d_in[11], *g1_b2 = (const float*)d_in[12];
  const float* g2_w1 = (const float*)d_in[13], *g2_b1 = (const float*)d_in[14];
  const float* g2_w2 = (const float*)d_in[15], *g2_b2 = (const float*)d_in[16];
  const float* f_w1  = (const float*)d_in[17], *f_b1  = (const float*)d_in[18];
  const float* f_w2  = (const float*)d_in[19], *f_b2  = (const float*)d_in[20];
  const float* o_w1  = (const float*)d_in[21], *o_b1  = (const float*)d_in[22];
  const float* o_w2  = (const float*)d_in[23], *o_b2  = (const float*)d_in[24];
  const float* o_w3  = (const float*)d_in[25], *o_b3  = (const float*)d_in[26];

  float* xsum = (float*)d_ws;     // [8192,32] f32 segment sums
  float* out  = (float*)d_out;    // [8192,2]

  k1_edge<<<8192, 256, 0, stream>>>(sn_feat, sn_id, graph_out,
                                    nn_w1, nn_b1, nn_w2, nn_b2,
                                    g1_w1, g1_b1, g1_w2, g1_b2, xsum);
  k2_head<<<32, 256, 0, stream>>>(xsum, s_feat,
                                  g2_w1, g2_b1, g2_w2, g2_b2,
                                  f_w1, f_b1, f_w2, f_b2,
                                  o_w1, o_b1, o_w2, o_b2, o_w3, o_b3, out);
}
